// ObjectosphereLoss_7971459301860
// MI455X (gfx1250) — matrix operand order repaired and checked by reading, not compiled, not run on hardware
//
#include <hip/hip_runtime.h>
#include <hip/hip_bf16.h>
#include <math.h>

// ObjectosphereLoss for MI455X (gfx1250, wave32).
//   B=262144 rows, D=512 features, C=13 classes, KNOWN=12, XI=10, lambda=1e-4.
// Memory-bound (512 MiB features -> ~23us @ 23.3 TB/s). Row norms computed on
// the WMMA path (Gram-tile diagonal, v_wmma_f32_16x16x32_f16), softmax on VALU.
// Hot loop is array-free; f16 packing via v_cvt_pk_rtz_f16_f32; unroll x4 for
// load/WMMA overlap without blowing VGPR pressure.

typedef __attribute__((ext_vector_type(16))) _Float16 v16h;
typedef __attribute__((ext_vector_type(8)))  float    v8f;
typedef __attribute__((ext_vector_type(8)))  int      v8i;

#define NB       262144
#define ND       512
#define NC       13
#define KNOWN_C  12
#define XI_C     10.0f
#define LAM_C    1e-4f
#define EPS_C    1e-10f

#define ROWS_PER_WAVE 16
#define WAVES_PER_BLK 8
#define ROWS_PER_BLK  (ROWS_PER_WAVE * WAVES_PER_BLK)   // 128
#define NBLOCKS1      (NB / ROWS_PER_BLK)               // 2048
#define KCHUNKS       (ND / 32)                         // 16

__device__ __forceinline__ int pk16(float lo, float hi) {
    auto r = __builtin_amdgcn_cvt_pkrtz(lo, hi);   // v_cvt_pk_rtz_f16_f32 (__fp16 x2)
    return __builtin_bit_cast(int, r);
}

__global__ __launch_bounds__(256) void obj_loss_main(
    const float* __restrict__ logits,        // [B, 13]
    const int*   __restrict__ true_y,        // [B]
    const int*   __restrict__ is_unknown,    // [B] (0/1)
    const float* __restrict__ feats,         // [B, 512]
    const float* __restrict__ weights,       // [13]
    float*       __restrict__ partials)      // [gridDim.x * 2] (num, den)
{
    __shared__ float ctile[WAVES_PER_BLK][16][17]; // Gram tiles (pad avoids bank conflicts)
    __shared__ float redsum[WAVES_PER_BLK][2];

    const int tid  = threadIdx.x;
    const int lane = tid & 31;
    const int wid  = tid >> 5;
    const int hi   = lane >> 4;          // 0 for lanes 0-15, 1 for lanes 16-31
    const int mrow = lane & 15;          // row-in-tile this lane services
    const int r0   = blockIdx.x * ROWS_PER_BLK + wid * ROWS_PER_WAVE;

    // ---- Phase 1: 16x16 Gram tile of the wave's 16 rows via WMMA ------------
    // Per lane, per 32-feature chunk of its row, load the 24-float window at
    // float-offset hi*8.  With d0..d11 = f16-pair dwords of that window:
    //   A (16x32 layout)  = d0..d3 ++ d8..d11                 (lane-invariant)
    //   B (32x16 layout)  = d0..d3 ++ d8..d11                 (same as A)
    v8f acc = {};
    const size_t winbase = (size_t)(r0 + mrow) * ND + (size_t)(hi * 8);

#pragma unroll 4
    for (int kc = 0; kc < KCHUNKS; ++kc) {
        const float4* p = (const float4*)(feats + winbase + (size_t)kc * 32);
        float4 f0 = p[0], f1 = p[1], f2 = p[2], f3 = p[3], f4 = p[4], f5 = p[5];

        const int d0  = pk16(f0.x, f0.y);
        const int d1  = pk16(f0.z, f0.w);
        const int d2  = pk16(f1.x, f1.y);
        const int d3  = pk16(f1.z, f1.w);
        const int d4  = pk16(f2.x, f2.y);
        const int d5  = pk16(f2.z, f2.w);
        const int d6  = pk16(f3.x, f3.y);
        const int d7  = pk16(f3.z, f3.w);
        const int d8  = pk16(f4.x, f4.y);
        const int d9  = pk16(f4.z, f4.w);
        const int d10 = pk16(f5.x, f5.y);
        const int d11 = pk16(f5.z, f5.w);

        v8i ai;
        ai[0] = d0;  ai[1] = d1;  ai[2] = d2;  ai[3] = d3;
        ai[4] = d8;  ai[5] = d9;  ai[6] = d10; ai[7] = d11;

        v8i bi;
        bi[0] = d0;
        bi[1] = d1;
        bi[2] = d2;
        bi[3] = d3;
        bi[4] = d8;
        bi[5] = d9;
        bi[6] = d10;
        bi[7] = d11;

        v16h a = __builtin_bit_cast(v16h, ai);
        v16h b = __builtin_bit_cast(v16h, bi);

        // D = A(16x32) x B(32x16) + C : Gram tile accumulation
        acc = __builtin_amdgcn_wmma_f32_16x16x32_f16(
            /*neg_a=*/false, a, /*neg_b=*/false, b,
            /*c_mod=*/(short)0, acc, /*reuse_a=*/false, /*reuse_b=*/false);
    }

    // C/D layout: VGPR r -> (M=r, N=lane) for lanes 0-15, (M=r+8, N=lane-16) else.
#pragma unroll
    for (int r = 0; r < 8; ++r) {
        ctile[wid][r + hi * 8][mrow] = acc[r];
    }
    __syncthreads();

    // ---- Phase 2: per-row softmax / loss (lanes 0-15, one row each) ---------
    float num = 0.0f, den = 0.0f;
    if (lane < 16) {
        const int   r    = r0 + mrow;
        const float n2   = ctile[wid][mrow][mrow];     // ||x_r||^2
        const float mag  = sqrtf(n2);

        const float* lp = logits + (size_t)r * NC;
        float lv[NC];
#pragma unroll
        for (int c = 0; c < NC; ++c) lv[c] = lp[c];

        float mx = lv[0];
#pragma unroll
        for (int c = 1; c < NC; ++c) mx = fmaxf(mx, lv[c]);

        float Z = 0.0f;
#pragma unroll
        for (int c = 0; c < NC; ++c) Z += __expf(lv[c] - mx);
        const float invZ = 1.0f / Z;

        int ty = true_y[r];
        ty = ty < 0 ? 0 : (ty > NC - 1 ? NC - 1 : ty);

        float sum_log = 0.0f, ls_ty = 0.0f;
#pragma unroll
        for (int c = 0; c < NC; ++c) {
            float sm = __expf(lv[c] - mx) * invZ;
            float ls = __logf(sm + EPS_C);
            sum_log += ls;
            if (c == ty) ls_ty = ls;                   // c compile-time -> select
        }

        const float je_un  = -(1.0f / (float)KNOWN_C) * sum_log;
        const float jr_un  = je_un + LAM_C * (mag * mag);
        const float pk     = fmaxf(XI_C - mag, 0.0f);
        const float jr_kn  = -ls_ty + LAM_C * (pk * pk);

        const float jr = (is_unknown[r] != 0) ? jr_un : jr_kn;
        const float w  = weights[ty];
        num = w * jr;
        den = w;
    }

    // ---- Phase 3: deterministic block reduction -----------------------------
#pragma unroll
    for (int off = 16; off > 0; off >>= 1) {
        num += __shfl_down(num, off, 32);
        den += __shfl_down(den, off, 32);
    }
    if (lane == 0) { redsum[wid][0] = num; redsum[wid][1] = den; }
    __syncthreads();
    if (tid == 0) {
        float n = 0.0f, d = 0.0f;
#pragma unroll
        for (int wv = 0; wv < WAVES_PER_BLK; ++wv) { n += redsum[wv][0]; d += redsum[wv][1]; }
        partials[2 * blockIdx.x]     = n;
        partials[2 * blockIdx.x + 1] = d;
    }
}

__global__ __launch_bounds__(256) void obj_loss_final(
    const float* __restrict__ partials, float* __restrict__ out, int nparts)
{
    __shared__ float s[8][2];
    const int tid  = threadIdx.x;
    const int lane = tid & 31;
    const int wid  = tid >> 5;

    float n = 0.0f, d = 0.0f;
    for (int i = tid; i < nparts; i += 256) {
        n += partials[2 * i];
        d += partials[2 * i + 1];
    }
#pragma unroll
    for (int off = 16; off > 0; off >>= 1) {
        n += __shfl_down(n, off, 32);
        d += __shfl_down(d, off, 32);
    }
    if (lane == 0) { s[wid][0] = n; s[wid][1] = d; }
    __syncthreads();
    if (tid == 0) {
        float nn = 0.0f, dd = 0.0f;
#pragma unroll
        for (int wv = 0; wv < 8; ++wv) { nn += s[wv][0]; dd += s[wv][1]; }
        out[0] = nn / dd;
    }
}

extern "C" void kernel_launch(void* const* d_in, const int* in_sizes, int n_in,
                              void* d_out, int out_size, void* d_ws, size_t ws_size,
                              hipStream_t stream) {
    const float* logits     = (const float*)d_in[0];
    const int*   true_y     = (const int*)  d_in[1];   // harness: integer -> int32
    const int*   is_unknown = (const int*)  d_in[2];   // bool promoted to int32
    const float* feats      = (const float*)d_in[3];
    const float* weights    = (const float*)d_in[4];

    float* partials = (float*)d_ws;                    // 2048 * 2 floats = 16 KiB

    obj_loss_main<<<NBLOCKS1, 256, 0, stream>>>(
        logits, true_y, is_unknown, feats, weights, partials);
    obj_loss_final<<<1, 256, 0, stream>>>(partials, (float*)d_out, NBLOCKS1);
}
